// GCN_LAYER_32753420599856
// MI455X (gfx1250) — matrix operand order repaired and checked by reading, not compiled, not run on hardware
//
#include <hip/hip_runtime.h>

// ---------------------------------------------------------------------------
// GCN layer for MI455X (gfx1250, wave32):
//   ah   = segment_sum(m, dst)                    [100000 x 24]
//   x    = [h | ah*norm]                          [100000 x 536]
//   y    = x @ W^T + b                            [100000 x 512]
//   out  = relu(layernorm(y) * gamma + beta)
//
// GEMM: V_WMMA_F32_16X16X32_BF16 with split-bf16 (hi+lo) fp32 emulation:
//   y ~= xh*Wh + xh*Wl + xl*Wh   (3 WMMAs/tile, ~1e-5 rel err)
// W streams once per WG via async-to-LDS double buffering (full K-block of
// latency cover).  LDS W layout interleaves hi/lo per 16-col group so every
// B-frag ds_load uses a 16-bit immediate offset from one per-lane base.
// Each 256-thread WG owns 64 nodes x all 512 outputs so LayerNorm+ReLU fuse
// into the epilogue.
// ---------------------------------------------------------------------------

typedef __attribute__((ext_vector_type(16))) __bf16         v16bf;
typedef __attribute__((ext_vector_type(8)))  float          v8f;
typedef __attribute__((ext_vector_type(16))) unsigned short v16u;
typedef __attribute__((ext_vector_type(8)))  unsigned short v8u;
typedef __attribute__((ext_vector_type(4)))  int            v4i;

#define AS1 __attribute__((address_space(1)))
#define AS3 __attribute__((address_space(3)))

#if __has_builtin(__builtin_amdgcn_global_load_async_to_lds_b128) && \
    __has_builtin(__builtin_amdgcn_s_wait_asynccnt)
#define USE_ASYNC 1
#else
#define USE_ASYNC 0
#endif

#define N_NODES 100000
#define N_EDGES 3200000
#define IN_F    512
#define E_F     24
#define OUT_F   512
#define KTOT    536           // IN_F + E_F
#define KPAD    544           // padded to 17 * 32
#define NKB     17            // K blocks of 32
#define LN_EPS  1e-5f

// LDS W buffer geometry: per 16-col group, hi block (16 cols x 36-elem
// stride = 576 elems) followed by lo block (576 elems) -> group stride 1152.
#define GSTRIDE 1152          // elements per 16-col group (hi+lo)
#define LOOFF   576           // lo block offset within group (elements)
#define WBUFSZ  (32 * GSTRIDE)  // 36864 elems = 73728 B per buffer

// float -> bf16 bits, round-to-nearest-even
__device__ __forceinline__ unsigned short f2bf(float f) {
  unsigned u = __float_as_uint(f);
  u += 0x7FFFu + ((u >> 16) & 1u);
  return (unsigned short)(u >> 16);
}
__device__ __forceinline__ float bf2f(unsigned short b) {
  return __uint_as_float(((unsigned)b) << 16);
}
__device__ __forceinline__ void split_bf16(float v, unsigned short& hi,
                                           unsigned short& lo) {
  hi = f2bf(v);
  lo = f2bf(v - bf2f(hi));
}

// assemble a v16bf WMMA fragment from two 8-element halves
__device__ __forceinline__ v16bf frag16(const unsigned short* p0,
                                        const unsigned short* p1) {
  v8u a = *(const v8u*)p0;
  v8u b = *(const v8u*)p1;
  v16u u;
#pragma unroll
  for (int j = 0; j < 8; ++j) { u[j] = a[j]; u[j + 8] = b[j]; }
  return __builtin_bit_cast(v16bf, u);
}

// ---------------------------------------------------------------------------
__global__ void k_zero_f32(float* __restrict__ p, int n) {
  int i = blockIdx.x * blockDim.x + threadIdx.x;
  if (i < n) p[i] = 0.f;
}

// Split W[512 x 536] fp32 into Wh/Wl bf16 in K-tiled layout:
//   Wx[(kb*OUT_F + col)*32 + klocal],  k padded to 544 with zeros.
__global__ void k_prep_w(const float* __restrict__ W,
                         unsigned short* __restrict__ Wh,
                         unsigned short* __restrict__ Wl) {
  int i = blockIdx.x * blockDim.x + threadIdx.x;
  if (i >= OUT_F * KPAD) return;
  int o = i / KPAD, k = i - o * KPAD;
  float v = (k < KTOT) ? W[o * KTOT + k] : 0.f;
  unsigned short hi, lo;
  split_bf16(v, hi, lo);
  int idx = (((k >> 5) * OUT_F) + o) * 32 + (k & 31);
  Wh[idx] = hi;
  Wl[idx] = lo;
}

// Segment sum: one thread per (edge, feat). m reads coalesced; fp32 global
// atomics scatter into ah (random dst, ~32 edges/node, L2 atomic units).
__global__ void k_seg_sum(const float* __restrict__ m,
                          const int* __restrict__ dst,
                          float* __restrict__ ah) {
  int i = blockIdx.x * blockDim.x + threadIdx.x;
  if (i >= N_EDGES * E_F) return;
  int e = i / E_F;
  int f = i - e * E_F;
  atomicAdd(&ah[dst[e] * E_F + f], m[i]);
}

// ---------------------------------------------------------------------------
// load one thread's 8-element slice of the x chunk for K-block kb
__device__ __forceinline__ void load_x_chunk(float v[8], int kb, int grow,
                                             int skl,
                                             const float* __restrict__ h,
                                             const float* __restrict__ ah,
                                             const float* __restrict__ norm) {
  if (kb < 16) {
    if (grow < N_NODES) {
      const float4* p =
          (const float4*)(h + (long long)grow * IN_F + kb * 32 + skl);
      float4 a = p[0], b4 = p[1];
      v[0] = a.x; v[1] = a.y; v[2] = a.z; v[3] = a.w;
      v[4] = b4.x; v[5] = b4.y; v[6] = b4.z; v[7] = b4.w;
    } else {
#pragma unroll
      for (int j = 0; j < 8; ++j) v[j] = 0.f;
    }
  } else {  // k in [512, 544): ah*norm for k<536, zero pad after
    float nv = (grow < N_NODES) ? norm[grow] : 0.f;
#pragma unroll
    for (int j = 0; j < 8; ++j) {
      int f = skl + j;
      v[j] = (f < E_F && grow < N_NODES) ? ah[grow * E_F + f] * nv : 0.f;
    }
  }
}

// Fused GEMM + bias + LayerNorm + ReLU.
// WG = 256 threads = 8 waves. Node tile: 64 rows x 512 cols.
// Wave w: rows [(w&3)*16, +16), cols [(w>>2)*256, +256) -> 16 accum tiles.
__global__ __launch_bounds__(256)
void k_gemm_ln(const float* __restrict__ h,
               const float* __restrict__ ah,
               const float* __restrict__ norm,
               const unsigned short* __restrict__ Wh,
               const unsigned short* __restrict__ Wl,
               const float* __restrict__ bias,
               const float* __restrict__ gamma,
               const float* __restrict__ beta,
               float* __restrict__ out) {
  __shared__ unsigned short XTh[64 * 36];  // x tile hi bf16, padded stride 36
  __shared__ unsigned short XTl[64 * 36];  // x tile lo bf16
  __shared__ float LNs[64];                // per-row sum(y)
  __shared__ float LNq[64];                // per-row sum(y^2)
#if USE_ASYNC
  // double-buffered W chunk, hi/lo interleaved per 16-col group
  __shared__ unsigned short WB[2][WBUFSZ];
#endif

  const int tid = threadIdx.x;
  const int R0  = blockIdx.x * 64;

  if (tid < 64)       LNs[tid]      = 0.f;
  else if (tid < 128) LNq[tid - 64] = 0.f;

  const int lane    = tid & 31;
  const int wave    = tid >> 5;
  const int rowbase = (wave & 3) * 16;
  const int colbase = (wave >> 2) * 256;
  const int lane_m  = lane & 15;   // M (rows of A) / N (cols of B) index
  const int halfL   = lane >> 4;   // selects K half (contiguous, k = 16*halfL + s)

  v8f acc[16] = {};

  // staging map: thread -> (row 0..63, klocal {0,8,16,24}), 8 elems each
  const int srow = tid >> 2;
  const int skl  = (tid & 3) * 8;
  const int grow = R0 + srow;

#if USE_ASYNC
  // async copy of W K-block kb into buffer b: 256 threads x (8+8) b128 ops,
  // lane-adjacent global addresses (fully coalesced); LDS dest is the
  // group-interleaved padded layout.
#define ASYNC_COPY_W(kb_, b_)                                               \
  {                                                                         \
    _Pragma("unroll") for (int i = 0; i < 8; ++i) {                         \
      int lin  = (i * 256 + tid) * 8; /* element index in 512x32 chunk */   \
      int col  = lin >> 5, kl = lin & 31;                                   \
      int loff = (col >> 4) * GSTRIDE + (col & 15) * 36 + kl;               \
      __builtin_amdgcn_global_load_async_to_lds_b128(                       \
          (AS1 v4i*)(Wh + (kb_) * (OUT_F * 32) + lin),                      \
          (AS3 v4i*)(&WB[(b_)][loff]), 0, 0);                               \
      __builtin_amdgcn_global_load_async_to_lds_b128(                       \
          (AS1 v4i*)(Wl + (kb_) * (OUT_F * 32) + lin),                      \
          (AS3 v4i*)(&WB[(b_)][loff + LOOFF]), 0, 0);                       \
    }                                                                       \
  }
  ASYNC_COPY_W(0, 0);
#endif

  float vnext[8];
  load_x_chunk(vnext, 0, grow, skl, h, ah, norm);

  for (int kb = 0; kb < NKB; ++kb) {
    // ---- stage x chunk [64 x 32] fp32 -> split bf16 in LDS ----
#pragma unroll
    for (int j = 0; j < 8; ++j) {
      unsigned short hi, lo;
      split_bf16(vnext[j], hi, lo);
      XTh[srow * 36 + skl + j] = hi;
      XTl[srow * 36 + skl + j] = lo;
    }

#if USE_ASYNC
    if (kb + 1 < NKB) {
      ASYNC_COPY_W(kb + 1, (kb + 1) & 1);          // prefetch next K-block
      __builtin_amdgcn_s_wait_asynccnt(16);        // current block landed
    } else {
      __builtin_amdgcn_s_wait_asynccnt(0);
    }
#endif
    __syncthreads();

    // overlap next x chunk's global loads with this block's WMMA work
    if (kb + 1 < NKB) load_x_chunk(vnext, kb + 1, grow, skl, h, ah, norm);

    // ---- A fragments (16x32 bf16, k = 16*halfL + s) from LDS ----
    const int arow = rowbase + lane_m;
    const unsigned short* pa = &XTh[arow * 36 + 16 * halfL];
    const unsigned short* pb = &XTl[arow * 36 + 16 * halfL];
    v16bf Afh = frag16(pa, pa + 8);
    v16bf Afl = frag16(pb, pb + 8);

    // ---- B fragment source ----
#if USE_ASYNC
    // per-lane base inside this kb's buffer; all 16 tiles reachable with
    // 16-bit immediate DS offsets (max 15*2304+1168+16 B < 64KB)
    const unsigned short* wbase =
        &WB[kb & 1][0] + (colbase >> 4) * GSTRIDE + lane_m * 36 + halfL * 16;
#define BH_PTR(t_) (wbase + (t_) * GSTRIDE)
#define BL_PTR(t_) (wbase + (t_) * GSTRIDE + LOOFF)
#else
    const unsigned short* bh0 =
        Wh + kb * (OUT_F * 32) + (colbase + lane_m) * 32 + halfL * 16;
    const unsigned short* bl0 =
        Wl + kb * (OUT_F * 32) + (colbase + lane_m) * 32 + halfL * 16;
#define BH_PTR(t_) (bh0 + (t_) * (16 * 32))
#define BL_PTR(t_) (bl0 + (t_) * (16 * 32))
#endif

    // one-tile-ahead software pipeline: loads of t+1 issue before WMMAs of t
    v16bf Bh = frag16(BH_PTR(0), BH_PTR(0) + 8);
    v16bf Bl = frag16(BL_PTR(0), BL_PTR(0) + 8);
#pragma unroll
    for (int t = 0; t < 16; ++t) {
      v16bf Bh2, Bl2;
      if (t < 15) {
        Bh2 = frag16(BH_PTR(t + 1), BH_PTR(t + 1) + 8);
        Bl2 = frag16(BL_PTR(t + 1), BL_PTR(t + 1) + 8);
      }
      acc[t] = __builtin_amdgcn_wmma_f32_16x16x32_bf16(
          false, Afh, false, Bh, (short)0, acc[t], false, false);
      acc[t] = __builtin_amdgcn_wmma_f32_16x16x32_bf16(
          false, Afh, false, Bl, (short)0, acc[t], false, false);
      acc[t] = __builtin_amdgcn_wmma_f32_16x16x32_bf16(
          false, Afl, false, Bh, (short)0, acc[t], false, false);
      if (t < 15) { Bh = Bh2; Bl = Bl2; }
    }
#undef BH_PTR
#undef BL_PTR
    __syncthreads();
  }

  // ---- epilogue: bias, LN stats, normalize, affine, relu, store ----
  float bcol[16], gcol[16], ecol[16];
#pragma unroll
  for (int t = 0; t < 16; ++t) {
    int c = colbase + t * 16 + lane_m;
    bcol[t] = bias[c];
    gcol[t] = gamma[c];
    ecol[t] = beta[c];
  }
#pragma unroll
  for (int t = 0; t < 16; ++t)
#pragma unroll
    for (int r = 0; r < 8; ++r) acc[t][r] += bcol[t];

#pragma unroll
  for (int r = 0; r < 8; ++r) {
    float s = 0.f, q = 0.f;
#pragma unroll
    for (int t = 0; t < 16; ++t) {
      float y = acc[t][r];
      s += y;
      q += y * y;
    }
    int lrow = rowbase + 8 * halfL + r;   // local node row this lane owns
    atomicAdd(&LNs[lrow], s);             // ds_add_f32 across 32 partials
    atomicAdd(&LNq[lrow], q);
  }
  __syncthreads();

#pragma unroll
  for (int r = 0; r < 8; ++r) {
    int lrow  = rowbase + 8 * halfL + r;
    int growo = R0 + lrow;
    if (growo >= N_NODES) continue;
    float mu   = LNs[lrow] * (1.f / OUT_F);
    float var  = LNq[lrow] * (1.f / OUT_F) - mu * mu;
    float rstd = rsqrtf(var + LN_EPS);
#pragma unroll
    for (int t = 0; t < 16; ++t) {
      float yv = (acc[t][r] - mu) * rstd * gcol[t] + ecol[t];
      out[(long long)growo * OUT_F + colbase + t * 16 + lane_m] =
          fmaxf(yv, 0.f);
    }
  }
}

// ---------------------------------------------------------------------------
extern "C" void kernel_launch(void* const* d_in, const int* in_sizes, int n_in,
                              void* d_out, int out_size, void* d_ws,
                              size_t ws_size, hipStream_t stream) {
  (void)in_sizes; (void)n_in; (void)out_size; (void)ws_size;

  const float* h     = (const float*)d_in[0];
  const float* m     = (const float*)d_in[1];
  const int*   dst   = (const int*)d_in[2];
  const float* norm  = (const float*)d_in[3];
  const float* W     = (const float*)d_in[4];
  const float* bias  = (const float*)d_in[5];
  const float* gamma = (const float*)d_in[6];
  const float* beta  = (const float*)d_in[7];
  float*       out   = (float*)d_out;

  // workspace layout
  unsigned short* Wh = (unsigned short*)d_ws;           // 17*512*32 bf16
  unsigned short* Wl = Wh + NKB * OUT_F * 32;           // 17*512*32 bf16
  float*          ab = (float*)(Wl + NKB * OUT_F * 32); // ah: 100000*24 f32

  // 1) zero ah accumulator
  {
    int n = N_NODES * E_F;
    k_zero_f32<<<(n + 255) / 256, 256, 0, stream>>>(ab, n);
  }
  // 2) split W into bf16 hi/lo, K-tiled layout
  {
    int n = OUT_F * KPAD;
    k_prep_w<<<(n + 255) / 256, 256, 0, stream>>>(W, Wh, Wl);
  }
  // 3) segment sum of edge features
  {
    int n = N_EDGES * E_F;
    k_seg_sum<<<(n + 255) / 256, 256, 0, stream>>>(m, dst, ab);
  }
  // 4) fused GEMM + LayerNorm + ReLU
  {
    int nblk = (N_NODES + 63) / 64;  // 1563
    k_gemm_ln<<<nblk, 256, 0, stream>>>(h, ab, norm, Wh, Wl, bias, gamma,
                                        beta, out);
  }
}
